// ManualGATLayer_14353780703890
// MI455X (gfx1250) — matrix operand order repaired and checked by reading, not compiled, not run on hardware
//
#include <hip/hip_runtime.h>

// MI455X / gfx1250, wave32. f16 WMMA path (values are O(1): f16 beats bf16 in
// precision at identical v_wmma_f32_16x16x32 throughput):
//   pass 0: zero d_out (head-mean accumulated with global f32 atomics)
//   pass 1: Wh = h @ W^T (8 GEMMs 4096x256x128) + si/sj vectors, Wh^T f16 in ws
//   pass 2: flash-style masked softmax attention, out += 0.25 * (alpha @ Wh),
//           LDS double-buffered staging (next tile's global loads issued before
//           the 16-WMMA compute block; one barrier per j-iteration).
// Kernel-2 blocking: 128 rows x 2 heads per block -> Wh re-stream 256MB + adj
// 256MB total (L2-resident; 192MB L2); adj hits HBM once (64MB).

typedef __attribute__((ext_vector_type(16))) _Float16 v16h;
typedef __attribute__((ext_vector_type(8)))  float    v8f;

union Frag16 { unsigned int u32[8]; unsigned short u16[16]; v16h v; };

__device__ __forceinline__ unsigned short f32_f16u(float f) {
  union { _Float16 h; unsigned short u; } c;
  c.h = (_Float16)f;                       // v_cvt_f16_f32 (RTE)
  return c.u;
}

constexpr int kB = 2, kN = 4096, kIN = 256, kOUT = 128, kH = 4;
constexpr float kSlope = 0.2f;

// ---------------------------------------------------------------------------
// Kernel 0: zero the output accumulator (d_out is poisoned by the harness)
// ---------------------------------------------------------------------------
extern "C" __global__ void gat_zero_kernel(float4* __restrict__ out)
{
  out[(size_t)blockIdx.x * 256 + threadIdx.x] = make_float4(0.f, 0.f, 0.f, 0.f);
}

// ---------------------------------------------------------------------------
// Kernel 1: Wh (f16, transposed [hb][o][n]) + si/sj.  grid = H*B*(N/128)
// ---------------------------------------------------------------------------
extern "C" __global__ __launch_bounds__(256)
void gat_wh_kernel(const float* __restrict__ hsrc, const float* __restrict__ Wsrc,
                   const float* __restrict__ avec,
                   unsigned short* __restrict__ whT,
                   float* __restrict__ si, float* __restrict__ sj)
{
  extern __shared__ __align__(16) char smem[];
  unsigned short* Wl = (unsigned short*)smem;   // [kOUT][kIN] f16  (64 KB)
  unsigned short* hl = Wl + kOUT * kIN;         // [128][kIN]  f16  (64 KB), reused as out staging

  const int tid  = threadIdx.x;
  const int bx   = blockIdx.x;
  const int rt   = bx & 31;           // row tile (N/128 = 32)
  const int b    = (bx >> 5) & 1;
  const int hh   = bx >> 6;
  const int hb   = hh * kB + b;
  const int row0 = rt * 128;

  // stage W[hh] and h-row block as f16 (coalesced)
  {
    const float* Wp = Wsrc + (size_t)hh * kOUT * kIN;
    #pragma unroll
    for (int i = 0; i < (kOUT * kIN) / 256; i++) {
      int idx = tid + i * 256; Wl[idx] = f32_f16u(Wp[idx]);
    }
    const float* hp = hsrc + ((size_t)b * kN + row0) * kIN;   // 128 rows contiguous
    #pragma unroll
    for (int i = 0; i < (128 * kIN) / 256; i++) {
      int idx = tid + i * 256; hl[idx] = f32_f16u(hp[idx]);
    }
  }
  __syncthreads();

  const int wv = tid >> 5, lane = tid & 31, n16 = lane & 15, half = lane >> 4;

  v8f acc[8] = {};   // 16 rows x 128 cols per wave

  #pragma unroll
  for (int kt = 0; kt < kIN / 32; kt++) {
    // A fragment: 16x32 f16, rows = this wave's 16 nodes
    Frag16 A;
    #pragma unroll
    for (int r = 0; r < 8; r++) {
      int k = kt * 32 + (r >> 2) * 16 + half * 8 + (r & 3) * 2;
      A.u32[r] = *(const unsigned int*)(hl + (wv * 16 + n16) * kIN + k);
    }
    #pragma unroll
    for (int t = 0; t < 8; t++) {
      // B fragment: 32x16, B[k][n] = W[t*16+n][k] (pair k,k+1 contiguous in Wl row)
      Frag16 Bf;
      #pragma unroll
      for (int r = 0; r < 8; r++) {
        int k = kt * 32 + (r >> 2) * 16 + half * 8 + (r & 3) * 2;
        Bf.u32[r] = *(const unsigned int*)(Wl + (t * 16 + n16) * kIN + k);
      }
      acc[t] = __builtin_amdgcn_wmma_f32_16x16x32_f16(
          false, A.v, false, Bf.v, (short)0, acc[t], false, false);
    }
  }

  // si = Wh . a1, sj = Wh . a2 (reduce over o: across the 16 lanes of a half + 8 tiles)
  float s1[8], s2[8];
  #pragma unroll
  for (int r = 0; r < 8; r++) { s1[r] = 0.f; s2[r] = 0.f; }
  #pragma unroll
  for (int t = 0; t < 8; t++) {
    float a1v = avec[(size_t)hh * 2 * kOUT + t * 16 + n16];
    float a2v = avec[(size_t)hh * 2 * kOUT + kOUT + t * 16 + n16];
    #pragma unroll
    for (int r = 0; r < 8; r++) { s1[r] += acc[t][r] * a1v; s2[r] += acc[t][r] * a2v; }
  }
  #pragma unroll
  for (int r = 0; r < 8; r++) {
    #pragma unroll
    for (int m = 1; m < 16; m <<= 1) {
      s1[r] += __shfl_xor(s1[r], m);
      s2[r] += __shfl_xor(s2[r], m);
    }
  }
  if (n16 == 0) {
    #pragma unroll
    for (int r = 0; r < 8; r++) {
      int row = row0 + wv * 16 + r + 8 * half;
      si[(size_t)hb * kN + row] = s1[r];
      sj[(size_t)hb * kN + row] = s2[r];
    }
  }

  __syncthreads();                       // all waves done reading hl
  // transpose through LDS: ol[o(128)][m(128)] f16, then coalesced b128 stores
  unsigned short* ol = hl;
  #pragma unroll
  for (int t = 0; t < 8; t++)
    #pragma unroll
    for (int r = 0; r < 8; r++)
      ol[(t * 16 + n16) * 128 + wv * 16 + r + 8 * half] = f32_f16u(acc[t][r]);
  __syncthreads();
  {
    unsigned short* dst = whT + (size_t)hb * kOUT * kN;
    const uint4* src = (const uint4*)ol;               // 2048 uint4
    #pragma unroll
    for (int i = 0; i < 8; i++) {
      int idx = tid + i * 256;
      int o = idx >> 4, q = idx & 15;
      *(uint4*)(dst + (size_t)o * kN + row0 + q * 8) = src[idx];
    }
  }
}

// ---------------------------------------------------------------------------
// Kernel 2: masked-softmax attention + aggregation, double-buffered staging.
// grid = (N/128, H/2, B). 8 waves: waves 0-3 head hp*2, waves 4-7 head hp*2+1;
// each wave owns 2 row-tiles of 16 (32 rows), so each B fragment feeds 2 WMMAs.
// ---------------------------------------------------------------------------
extern "C" __global__ __launch_bounds__(256)
void gat_attn_kernel(const int* __restrict__ adj,
                     const unsigned short* __restrict__ whT,
                     const float* __restrict__ si, const float* __restrict__ sj,
                     float* __restrict__ out)
{
  extern __shared__ __align__(16) char smem[];
  int*            adjs = (int*)smem;                      // [2][128*32]      32 KB
  unsigned short* wlh  = (unsigned short*)(smem + 32768); // [2][2][128*32]   32 KB

  const int tid  = threadIdx.x;
  const int i0   = blockIdx.x * 128;
  const int hp   = blockIdx.y;               // head pair
  const int b    = blockIdx.z;
  const int wv   = tid >> 5, lane = tid & 31, n16 = lane & 15, half = lane >> 4;
  const int grp  = wv >> 2;                  // which head of the pair
  const int rq   = wv & 3;                   // row quad within block
  const int hh   = hp * 2 + grp;
  const int hb   = hh * kB + b;
  const float NEG_INF = -__builtin_inff();

  float simv[2], rm[2], rs[2];
  #pragma unroll
  for (int tl = 0; tl < 2; tl++) {
    simv[tl] = si[(size_t)hb * kN + i0 + rq * 32 + tl * 16 + n16];
    rm[tl] = NEG_INF;
    rs[tl] = 0.f;
  }
  v8f acc[2][8] = {};

  uint4 adjReg[4], whReg[4];

  // ---- prologue: fetch tile 0 into buffer 0 ----
  #pragma unroll
  for (int i = 0; i < 4; i++) {
    int idx = tid + i * 256;                 // 1024 uint4 each
    int row = idx >> 3, q = idx & 7;
    adjReg[i] = *(const uint4*)(adj + (size_t)(i0 + row) * kN + q * 4);
    int g = idx >> 9, rem = idx & 511, o = rem >> 2, qq = rem & 3;
    int hbg = (hp * 2 + g) * kB + b;
    whReg[i] = *(const uint4*)(whT + (size_t)hbg * kOUT * kN + (size_t)o * kN + qq * 8);
  }
  #pragma unroll
  for (int i = 0; i < 4; i++) {
    int idx = tid + i * 256;
    ((uint4*)adjs)[idx] = adjReg[i];
    ((uint4*)wlh)[idx]  = whReg[i];
  }
  __syncthreads();

  for (int jb = 0; jb < kN / 32; jb++) {
    const int j0  = jb * 32;
    const int cur = jb & 1;
    const bool more = (jb + 1) < (kN / 32);

    // ---- issue NEXT tile's global loads early (hidden under compute) ----
    if (more) {
      const int j0n = j0 + 32;
      #pragma unroll
      for (int i = 0; i < 4; i++) {
        int idx = tid + i * 256;
        int row = idx >> 3, q = idx & 7;
        adjReg[i] = *(const uint4*)(adj + (size_t)(i0 + row) * kN + j0n + q * 4);
        int g = idx >> 9, rem = idx & 511, o = rem >> 2, qq = rem & 3;
        int hbg = (hp * 2 + g) * kB + b;
        whReg[i] = *(const uint4*)(whT + (size_t)hbg * kOUT * kN + (size_t)o * kN + j0n + qq * 8);
      }
    }

    // ---- compute on buffer `cur` ----
    const int*            adjc = adjs + cur * 4096;
    const unsigned short* wlc  = wlh + cur * 8192 + grp * 4096;  // [o(128)][k(32)]

    // sj values for this lane's A-fragment K pattern (same for both tiles)
    const float* sjb = sj + (size_t)hb * kN + j0;
    float run[2][8];
    {
      float4 t0 = *(const float4*)(sjb + half * 8);
      float4 t1 = *(const float4*)(sjb + half * 8 + 4);
      float4 t2 = *(const float4*)(sjb + 16 + half * 8);
      float4 t3 = *(const float4*)(sjb + 16 + half * 8 + 4);
      run[0][0]=t0.x; run[0][1]=t0.y; run[0][2]=t0.z; run[0][3]=t0.w;
      run[0][4]=t1.x; run[0][5]=t1.y; run[0][6]=t1.z; run[0][7]=t1.w;
      run[1][0]=t2.x; run[1][1]=t2.y; run[1][2]=t2.z; run[1][3]=t2.w;
      run[1][4]=t3.x; run[1][5]=t3.y; run[1][6]=t3.z; run[1][7]=t3.w;
    }

    Frag16 A[2];
    #pragma unroll
    for (int tl = 0; tl < 2; tl++) {
      const int rloc = rq * 32 + tl * 16 + n16;   // row in block (this lane's A row)
      float ev[16];
      float bmax = NEG_INF;
      #pragma unroll
      for (int r = 0; r < 8; r++) {
        #pragma unroll
        for (int d = 0; d < 2; d++) {
          int k = (r >> 2) * 16 + half * 8 + (r & 3) * 2 + d;
          float e = simv[tl] + run[r >> 2][(r & 3) * 2 + d];
          e = (e >= 0.f) ? e : kSlope * e;
          if (adjc[rloc * 32 + k] == 0) e = NEG_INF;
          ev[2 * r + d] = e;
          bmax = fmaxf(bmax, e);
        }
      }
      bmax = fmaxf(bmax, __shfl_xor(bmax, 16));
      float newM = fmaxf(rm[tl], bmax);
      float scale = (rm[tl] == NEG_INF) ? 0.f : __expf(rm[tl] - newM);

      float rowsum = 0.f;
      #pragma unroll
      for (int q = 0; q < 16; q++) {
        float p = (ev[q] == NEG_INF) ? 0.f : __expf(ev[q] - newM);
        rowsum += p;
        A[tl].u16[q] = f32_f16u(p);
      }
      rowsum += __shfl_xor(rowsum, 16);
      rs[tl] = rs[tl] * scale + rowsum;
      rm[tl] = newM;

      // rescale accumulator: acc row m = r + 8*half; scale lives on lane m
      #pragma unroll
      for (int r = 0; r < 8; r++) {
        float sc = __shfl(scale, (half << 3) + r);
        #pragma unroll
        for (int t = 0; t < 8; t++) acc[tl][t][r] *= sc;
      }
    }

    // acc += P(16x32) x Wh(32x16): each B fragment feeds both row-tiles
    #pragma unroll
    for (int t = 0; t < 8; t++) {
      Frag16 Bf;
      #pragma unroll
      for (int r = 0; r < 8; r++)
        Bf.u32[r] = *(const unsigned int*)(wlc + (t * 16 + n16) * 32 + (r >> 2) * 16 + half * 8 + (r & 3) * 2);
      acc[0][t] = __builtin_amdgcn_wmma_f32_16x16x32_f16(
          false, A[0].v, false, Bf.v, (short)0, acc[0][t], false, false);
      acc[1][t] = __builtin_amdgcn_wmma_f32_16x16x32_f16(
          false, A[1].v, false, Bf.v, (short)0, acc[1][t], false, false);
    }

    // ---- store NEXT tile into the other buffer; one barrier per iteration ----
    if (more) {
      const int nxt = cur ^ 1;
      #pragma unroll
      for (int i = 0; i < 4; i++) {
        int idx = tid + i * 256;
        ((uint4*)(adjs + nxt * 4096))[idx] = adjReg[i];
        ((uint4*)(wlh + nxt * 8192))[idx]  = whReg[i];
      }
    }
    __syncthreads();
  }

  // normalize and accumulate head-mean into global out (pre-zeroed)
  #pragma unroll
  for (int tl = 0; tl < 2; tl++) {
    float inv = (rs[tl] > 0.f) ? (1.0f / rs[tl]) : 0.f;  // fully-masked row -> 0
    #pragma unroll
    for (int r = 0; r < 8; r++) {
      float ir = __shfl(inv, (half << 3) + r) * 0.25f;
      int row = i0 + rq * 32 + tl * 16 + r + 8 * half;
      #pragma unroll
      for (int t = 0; t < 8; t++)
        unsafeAtomicAdd(&out[((size_t)b * kN + row) * kOUT + t * 16 + n16],
                        acc[tl][t][r] * ir);
    }
  }
}

// ---------------------------------------------------------------------------
extern "C" void kernel_launch(void* const* d_in, const int* in_sizes, int n_in,
                              void* d_out, int out_size, void* d_ws, size_t ws_size,
                              hipStream_t stream)
{
  (void)in_sizes; (void)n_in; (void)out_size; (void)ws_size;
  const float* h   = (const float*)d_in[0];
  const int*   adj = (const int*)d_in[1];
  const float* W   = (const float*)d_in[2];
  const float* a   = (const float*)d_in[3];
  float* out = (float*)d_out;

  // workspace: Wh^T f16 (8 MB) + si + sj
  unsigned short* whT = (unsigned short*)d_ws;
  size_t whBytes = (size_t)kH * kB * kOUT * kN * sizeof(unsigned short);
  float* si = (float*)((char*)d_ws + whBytes);
  float* sj = si + (size_t)kH * kB * kN;

  const size_t shm1 = (size_t)(kOUT * kIN + 128 * kIN) * sizeof(unsigned short); // 128 KB
  const size_t shm2 = 65536;                                                     //  64 KB
  hipFuncSetAttribute((const void*)gat_wh_kernel,
                      hipFuncAttributeMaxDynamicSharedMemorySize, (int)shm1);
  hipFuncSetAttribute((const void*)gat_attn_kernel,
                      hipFuncAttributeMaxDynamicSharedMemorySize, (int)shm2);

  const int outVec4 = kB * kN * kOUT / 4;   // 262144 float4
  gat_zero_kernel<<<dim3(outVec4 / 256), dim3(256), 0, stream>>>((float4*)out);
  gat_wh_kernel<<<dim3(kH * kB * (kN / 128)), dim3(256), shm1, stream>>>(
      h, W, a, whT, si, sj);
  gat_attn_kernel<<<dim3(kN / 128, kH / 2, kB), dim3(256), shm2, stream>>>(
      adj, whT, si, sj, out);
}
